// TernaryLinear_21423296872678
// MI455X (gfx1250) — compile-verified, matrix operand order checked by reading, not run on hardware
//
#include <hip/hip_runtime.h>

// ---- WMMA vector types (CDNA5 / gfx1250, wave32) ----
typedef __attribute__((ext_vector_type(16))) __bf16         v16bf;
typedef __attribute__((ext_vector_type(8)))  float          v8f;
typedef __attribute__((ext_vector_type(8)))  unsigned short ushort8;
typedef __attribute__((ext_vector_type(4)))  unsigned short ushort4v;
typedef __attribute__((ext_vector_type(4)))  int            v4i;

#define BM 128
#define BN 128
#define BK 32
#define LDA 40   // padded LDS row stride (elements): 80 B -> conflict-free 16-lane frag reads

#if defined(__gfx1250__) && __has_builtin(__builtin_amdgcn_global_load_async_to_lds_b128)
#define HAVE_ASYNC_LDS 1
#else
#define HAVE_ASYNC_LDS 0
#endif

union Frag16 {
    ushort8 h[2];
    v16bf   v;
};

// float -> bf16 bits, round-to-nearest-even
__device__ __forceinline__ unsigned short f2bf_rne(float f) {
    unsigned u = __builtin_bit_cast(unsigned, f);
    return (unsigned short)((u + 0x7FFFu + ((u >> 16) & 1u)) >> 16);
}

// 16-byte global -> LDS copy: async DMA pipe when available (no VGPR relay)
__device__ __forceinline__ void copy16_g2l(const unsigned short* g, unsigned short* l) {
#if HAVE_ASYNC_LDS
    __builtin_amdgcn_global_load_async_to_lds_b128(
        (__attribute__((address_space(1))) v4i*)g,
        (__attribute__((address_space(3))) v4i*)l,
        0, 0);
#else
    *reinterpret_cast<ushort8*>(l) = *reinterpret_cast<const ushort8*>(g);
#endif
}

__device__ __forceinline__ void wait_async_copies() {
#if HAVE_ASYNC_LDS
#if __has_builtin(__builtin_amdgcn_s_wait_asynccnt)
    __builtin_amdgcn_s_wait_asynccnt(0);
#else
    asm volatile("s_wait_asynccnt 0x0" ::: "memory");
#endif
#endif
}

// ============================================================================
// Phase 1a: split x (f32) -> x_hi + x_lo (bf16 each); exact to ~16 mantissa bits
// ============================================================================
__global__ __launch_bounds__(256)
void prep_x_split(const float* __restrict__ X,
                  unsigned short* __restrict__ Xhi,
                  unsigned short* __restrict__ Xlo) {
    size_t base = ((size_t)blockIdx.x * 256 + threadIdx.x) * 8;
    const float4 a = *reinterpret_cast<const float4*>(&X[base]);
    const float4 b = *reinterpret_cast<const float4*>(&X[base + 4]);
    float xs[8] = {a.x, a.y, a.z, a.w, b.x, b.y, b.z, b.w};
    ushort8 hv, lv;
    #pragma unroll
    for (int e = 0; e < 8; ++e) {
        unsigned short hi = f2bf_rne(xs[e]);
        float hif = __builtin_bit_cast(float, (unsigned)hi << 16);
        hv[e] = hi;
        lv[e] = f2bf_rne(xs[e] - hif);
    }
    *reinterpret_cast<ushort8*>(&Xhi[base]) = hv;
    *reinterpret_cast<ushort8*>(&Xlo[base]) = lv;
}

// ============================================================================
// Phase 1b: ternarize W (f32) -> {-1,0,+1} as bf16 bits (exactly representable)
// ============================================================================
__global__ __launch_bounds__(256)
void prep_w_ternary(const float* __restrict__ W,
                    unsigned short* __restrict__ Wq) {
    size_t base = ((size_t)blockIdx.x * 256 + threadIdx.x) * 8;
    const float4 a = *reinterpret_cast<const float4*>(&W[base]);
    const float4 b = *reinterpret_cast<const float4*>(&W[base + 4]);
    float ws[8] = {a.x, a.y, a.z, a.w, b.x, b.y, b.z, b.w};
    ushort8 qv;
    #pragma unroll
    for (int e = 0; e < 8; ++e) {
        float w = ws[e];
        qv[e] = (fabsf(w) > 0.5f) ? (w > 0.0f ? (unsigned short)0x3F80u
                                              : (unsigned short)0xBF80u)
                                  : (unsigned short)0u;
    }
    *reinterpret_cast<ushort8*>(&Wq[base]) = qv;
}

// stage one 128x32 bf16 tile triple into LDS (6 x b128 per thread)
__device__ __forceinline__ void stage_tile(const unsigned short* __restrict__ Xhi,
                                           const unsigned short* __restrict__ Xlo,
                                           const unsigned short* __restrict__ Wq,
                                           unsigned short* sAhi, unsigned short* sAlo,
                                           unsigned short* sB,
                                           int tid, int m0, int n0, int K, int kb) {
    #pragma unroll
    for (int it = 0; it < 2; ++it) {
        int g = tid + it * 256, row = g >> 2, c8 = (g & 3) * 8;
        size_t xo = (size_t)(m0 + row) * K + kb + c8;
        size_t wo = (size_t)(n0 + row) * K + kb + c8;
        int lo = row * LDA + c8;
        copy16_g2l(&Xhi[xo], &sAhi[lo]);
        copy16_g2l(&Xlo[xo], &sAlo[lo]);
        copy16_g2l(&Wq[wo],  &sB[lo]);
    }
}

// ============================================================================
// Phase 2: bf16 WMMA GEMM, double-buffered LDS, async global->LDS staging
//   y[m,n] = sum_k (xhi+xlo)[m,k] * wq[n,k]   (f32 accumulate on WMMA)
// ============================================================================
__global__ __launch_bounds__(256)
void ternary_gemm_wmma_pre(const unsigned short* __restrict__ Xhi,
                           const unsigned short* __restrict__ Xlo,
                           const unsigned short* __restrict__ Wq,
                           float* __restrict__ Y, int M, int N, int K) {
    __shared__ __align__(16) unsigned short sAhi[2][BM * LDA];
    __shared__ __align__(16) unsigned short sAlo[2][BM * LDA];
    __shared__ __align__(16) unsigned short sB  [2][BN * LDA];

    const int tid  = threadIdx.x;
    const int lane = tid & 31;
    const int wave = tid >> 5;
    const int wm   = wave & 3;    // 4 waves along M: 32 rows each
    const int wn   = wave >> 2;   // 2 waves along N: 64 cols each

    const int m0 = blockIdx.y * BM;
    const int n0 = blockIdx.x * BN;

    v8f acc[2][4] = {};
    const int koff0 = (lane >> 4) * 8;  // lanes<16: K 0..7/16..23; lanes>=16: 8..15/24..31

    // ---- prologue: stage tile kb=0 into buffer 0 ----
    stage_tile(Xhi, Xlo, Wq, &sAhi[0][0], &sAlo[0][0], &sB[0][0], tid, m0, n0, K, 0);
    wait_async_copies();
    __syncthreads();

    int cur = 0;
    for (int kb = 0; kb < K; kb += BK) {
        // ---- kick off next tile's async global->LDS copies (hide under WMMAs) ----
        const bool has_next = (kb + BK) < K;
        if (has_next) {
            stage_tile(Xhi, Xlo, Wq, &sAhi[cur ^ 1][0], &sAlo[cur ^ 1][0],
                       &sB[cur ^ 1][0], tid, m0, n0, K, kb + BK);
        }

        // ---- compute from buffer `cur` ----
        Frag16 bfrag[4];
        #pragma unroll
        for (int j = 0; j < 4; ++j) {
            int nrow = wn * 64 + j * 16 + (lane & 15);
            bfrag[j].h[0] = *reinterpret_cast<const ushort8*>(&sB[cur][nrow * LDA + koff0]);
            bfrag[j].h[1] = *reinterpret_cast<const ushort8*>(&sB[cur][nrow * LDA + koff0 + 16]);
        }
        #pragma unroll
        for (int i = 0; i < 2; ++i) {
            int mrow = wm * 32 + i * 16 + (lane & 15);
            Frag16 ahi, alo;
            ahi.h[0] = *reinterpret_cast<const ushort8*>(&sAhi[cur][mrow * LDA + koff0]);
            ahi.h[1] = *reinterpret_cast<const ushort8*>(&sAhi[cur][mrow * LDA + koff0 + 16]);
            alo.h[0] = *reinterpret_cast<const ushort8*>(&sAlo[cur][mrow * LDA + koff0]);
            alo.h[1] = *reinterpret_cast<const ushort8*>(&sAlo[cur][mrow * LDA + koff0 + 16]);
            #pragma unroll
            for (int j = 0; j < 4; ++j) {
                acc[i][j] = __builtin_amdgcn_wmma_f32_16x16x32_bf16(
                    false, ahi.v, false, bfrag[j].v, (short)0, acc[i][j], false, false);
                acc[i][j] = __builtin_amdgcn_wmma_f32_16x16x32_bf16(
                    false, alo.v, false, bfrag[j].v, (short)0, acc[i][j], false, false);
            }
        }

        // ---- drain this thread's async copies, then make LDS visible to all ----
        if (has_next) wait_async_copies();
        __syncthreads();
        cur ^= 1;
    }

    // ---- store (non-temporal: keep L2 for the resident bf16 panels) ----
    const int rsel = (lane >> 4) * 8;
    #pragma unroll
    for (int i = 0; i < 2; ++i) {
        #pragma unroll
        for (int j = 0; j < 4; ++j) {
            int col  = n0 + wn * 64 + j * 16 + (lane & 15);
            int rowb = m0 + wm * 32 + i * 16 + rsel;
            #pragma unroll
            for (int v = 0; v < 8; ++v) {
                __builtin_nontemporal_store(acc[i][j][v], &Y[(size_t)(rowb + v) * N + col]);
            }
        }
    }
}

// ============================================================================
// Fallback: convert-in-kernel GEMM (used only if d_ws is too small)
// ============================================================================
__global__ __launch_bounds__(256)
void ternary_gemm_wmma(const float* __restrict__ X, const float* __restrict__ W,
                       float* __restrict__ Y, int M, int N, int K) {
    __shared__ __align__(16) unsigned short sAhi[BM * LDA];
    __shared__ __align__(16) unsigned short sAlo[BM * LDA];
    __shared__ __align__(16) unsigned short sB  [BN * LDA];

    const int tid  = threadIdx.x;
    const int lane = tid & 31;
    const int wave = tid >> 5;
    const int wm   = wave & 3;
    const int wn   = wave >> 2;
    const int m0 = blockIdx.y * BM;
    const int n0 = blockIdx.x * BN;

    v8f acc[2][4] = {};
    const int koff0 = (lane >> 4) * 8;

    for (int kb = 0; kb < K; kb += BK) {
        #pragma unroll
        for (int it = 0; it < 4; ++it) {
            int idx = tid + it * 256;
            int row = idx >> 3;
            int kq  = (idx & 7) * 4;
            const float4 xv = *reinterpret_cast<const float4*>(
                &X[(size_t)(m0 + row) * K + kb + kq]);
            float xs[4] = {xv.x, xv.y, xv.z, xv.w};
            ushort4v hv, lv;
            #pragma unroll
            for (int e = 0; e < 4; ++e) {
                unsigned short hi = f2bf_rne(xs[e]);
                float hif = __builtin_bit_cast(float, (unsigned)hi << 16);
                hv[e] = hi;
                lv[e] = f2bf_rne(xs[e] - hif);
            }
            *reinterpret_cast<ushort4v*>(&sAhi[row * LDA + kq]) = hv;
            *reinterpret_cast<ushort4v*>(&sAlo[row * LDA + kq]) = lv;

            const float4 wv = *reinterpret_cast<const float4*>(
                &W[(size_t)(n0 + row) * K + kb + kq]);
            float wsv[4] = {wv.x, wv.y, wv.z, wv.w};
            ushort4v qv;
            #pragma unroll
            for (int e = 0; e < 4; ++e) {
                float w = wsv[e];
                qv[e] = (fabsf(w) > 0.5f) ? (w > 0.0f ? (unsigned short)0x3F80u
                                                      : (unsigned short)0xBF80u)
                                          : (unsigned short)0u;
            }
            *reinterpret_cast<ushort4v*>(&sB[row * LDA + kq]) = qv;
        }
        __syncthreads();

        if (kb + BK < K) {
            int prow = tid >> 1;
            int pcol = kb + BK + (tid & 1) * 16;
            __builtin_prefetch(&X[(size_t)(m0 + prow) * K + pcol], 0, 3);
            __builtin_prefetch(&W[(size_t)(n0 + prow) * K + pcol], 0, 3);
        }

        Frag16 bfrag[4];
        #pragma unroll
        for (int j = 0; j < 4; ++j) {
            int nrow = wn * 64 + j * 16 + (lane & 15);
            bfrag[j].h[0] = *reinterpret_cast<const ushort8*>(&sB[nrow * LDA + koff0]);
            bfrag[j].h[1] = *reinterpret_cast<const ushort8*>(&sB[nrow * LDA + koff0 + 16]);
        }
        #pragma unroll
        for (int i = 0; i < 2; ++i) {
            int mrow = wm * 32 + i * 16 + (lane & 15);
            Frag16 ahi, alo;
            ahi.h[0] = *reinterpret_cast<const ushort8*>(&sAhi[mrow * LDA + koff0]);
            ahi.h[1] = *reinterpret_cast<const ushort8*>(&sAhi[mrow * LDA + koff0 + 16]);
            alo.h[0] = *reinterpret_cast<const ushort8*>(&sAlo[mrow * LDA + koff0]);
            alo.h[1] = *reinterpret_cast<const ushort8*>(&sAlo[mrow * LDA + koff0 + 16]);
            #pragma unroll
            for (int j = 0; j < 4; ++j) {
                acc[i][j] = __builtin_amdgcn_wmma_f32_16x16x32_bf16(
                    false, ahi.v, false, bfrag[j].v, (short)0, acc[i][j], false, false);
                acc[i][j] = __builtin_amdgcn_wmma_f32_16x16x32_bf16(
                    false, alo.v, false, bfrag[j].v, (short)0, acc[i][j], false, false);
            }
        }
        __syncthreads();
    }

    const int rsel = (lane >> 4) * 8;
    #pragma unroll
    for (int i = 0; i < 2; ++i) {
        #pragma unroll
        for (int j = 0; j < 4; ++j) {
            int col  = n0 + wn * 64 + j * 16 + (lane & 15);
            int rowb = m0 + wm * 32 + i * 16 + rsel;
            #pragma unroll
            for (int v = 0; v < 8; ++v) {
                __builtin_nontemporal_store(acc[i][j][v], &Y[(size_t)(rowb + v) * N + col]);
            }
        }
    }
}

extern "C" void kernel_launch(void* const* d_in, const int* in_sizes, int n_in,
                              void* d_out, int out_size, void* d_ws, size_t ws_size,
                              hipStream_t stream) {
    (void)n_in; (void)out_size;
    const float* X = (const float*)d_in[0];   // [M,K]
    const float* W = (const float*)d_in[1];   // [N,K]
    float* Y = (float*)d_out;                 // [M,N]

    const int K = 4096;
    const int M = in_sizes[0] / K;            // 8192
    const int N = in_sizes[1] / K;            // 4096

    dim3 grid(N / BN, M / BM);                // (32, 64)

    const size_t xElems = (size_t)M * K;
    const size_t wElems = (size_t)N * K;
    const size_t need   = xElems * 2 * 2 + wElems * 2;  // xhi + xlo + wq (bf16)

    if (ws_size >= need) {
        unsigned short* xhi = (unsigned short*)d_ws;
        unsigned short* xlo = xhi + xElems;
        unsigned short* wq  = xlo + xElems;
        prep_x_split  <<<dim3((unsigned)(xElems / 8 / 256)), 256, 0, stream>>>(X, xhi, xlo);
        prep_w_ternary<<<dim3((unsigned)(wElems / 8 / 256)), 256, 0, stream>>>(W, wq);
        ternary_gemm_wmma_pre<<<grid, 256, 0, stream>>>(xhi, xlo, wq, Y, M, N, K);
    } else {
        ternary_gemm_wmma<<<grid, 256, 0, stream>>>(X, W, Y, M, N, K);
    }
}
